// NLSA_12369505812980
// MI455X (gfx1250) — compile-verified, matrix operand order checked by reading, not run on hardware
//
#include <hip/hip_runtime.h>
#include <hip/hip_bf16.h>

// Problem constants (from reference setup_inputs)
#define BB 8
#define CC 256
#define HH 96
#define WW 96
#define SS (HH * WW)      // 9216
#define AA 4
#define MM 64             // projection dim
#define KB 144            // bucket size
#define NB (SS / KB)      // 64 buckets
#define KPAD 160          // KB padded to multiple of 32 for WMMA K loop

typedef __bf16 v16bf __attribute__((ext_vector_type(16)));
typedef float  v8f   __attribute__((ext_vector_type(8)));
typedef unsigned int v4u __attribute__((ext_vector_type(4)));
typedef int v8i __attribute__((ext_vector_type(8)));
typedef int v4i __attribute__((ext_vector_type(4)));

// ---------------------------------------------------------------------------
// TDM: async-load a contiguous run of `nelems` f32 (nelems < 65536) from
// global to LDS. Descriptor layout per CDNA5 ISA:
// 1-row tensor, data_size=4B, tile_dim0=nelems, type=2 ("image").
// Issue from one wave only; completion via s_wait_tensorcnt.
// This toolchain uses the 6-arg builtin (g0, g1, g2, g3, g4, cpol).
// ---------------------------------------------------------------------------
__device__ __forceinline__ void tdm_load_f32(unsigned lds_off,
                                             const void* gptr,
                                             unsigned nelems) {
    unsigned long long ga = (unsigned long long)(uintptr_t)gptr;
    v4u g0;
    g0[0] = 1u;                                        // count=1, user desc
    g0[1] = lds_off;                                   // lds_addr (bytes)
    g0[2] = (unsigned)(ga & 0xffffffffu);              // global_addr[31:0]
    g0[3] = (unsigned)((ga >> 32) & 0x01ffffffu)       // global_addr[56:32]
          | (2u << 30);                                // type=2
    v8i g1;
    g1[0] = 0x00020000;                                // data_size=2 (4B)
    g1[1] = (int)((nelems & 0xffffu) << 16);           // tensor_dim0[15:0]
    g1[2] = (int)(((nelems >> 16) & 0xffffu)           // tensor_dim0[31:16]
          | (1u << 16));                               // tensor_dim1 = 1
    g1[3] = (int)((nelems & 0xffffu) << 16);           // tile_dim0
    g1[4] = 0;                                         // tile_dim1/2 unused
    g1[5] = (int)nelems;                               // dim0_stride[31:0]
    g1[6] = 0;                                         // strides hi
    g1[7] = 0;
    v4i z4 = {0, 0, 0, 0};
    v8i z8 = {0, 0, 0, 0, 0, 0, 0, 0};
    __builtin_amdgcn_tensor_load_to_lds(g0, g1, z4, z4, z8, 0);
}

// ---------------------------------------------------------------------------
// Kernel 0: zero the output accumulator
// ---------------------------------------------------------------------------
__global__ void nlsa_zero(float* __restrict__ out, int n) {
    int i = blockIdx.x * blockDim.x + threadIdx.x;
    if (i < n) out[i] = 0.0f;
}

// ---------------------------------------------------------------------------
// Kernel 1: LSH projection + max-hash.
// hash[b][a][s] = max_m sum_c in[b][c][s] * R[a][c][m]
// R staged in 64x64 LDS chunks via the Tensor Data Mover.
// ---------------------------------------------------------------------------
__global__ void nlsa_hash(const float* __restrict__ in,
                          const float* __restrict__ rm,
                          float* __restrict__ hash) {
    __shared__ float Rs[64][MM];
    const int tiles = SS / 256;            // 36
    int ba   = blockIdx.x / tiles;
    int tile = blockIdx.x % tiles;
    int b = ba / AA;
    int a = ba % AA;
    int s = tile * 256 + threadIdx.x;

    float acc[MM];
#pragma unroll
    for (int m = 0; m < MM; ++m) acc[m] = 0.0f;

    for (int ct = 0; ct < CC / 64; ++ct) {
        __syncthreads();                   // previous chunk fully consumed
        if (threadIdx.x < 32) {            // one wave issues the TDM op
            tdm_load_f32((unsigned)(uintptr_t)(void*)Rs,
                         &rm[((a * CC) + ct * 64) * MM], 64 * MM);
            __builtin_amdgcn_s_wait_tensorcnt(0);
        }
        __syncthreads();
        for (int cl = 0; cl < 64; ++cl) {
            float xc = in[((b * CC) + ct * 64 + cl) * SS + s];
#pragma unroll
            for (int m = 0; m < MM; ++m)
                acc[m] = fmaf(xc, Rs[cl][m], acc[m]);
        }
    }
    float mx = acc[0];
#pragma unroll
    for (int m = 1; m < MM; ++m) mx = fmaxf(mx, acc[m]);
    hash[ba * SS + s] = mx;
}

// ---------------------------------------------------------------------------
// Kernel 2: stable descending rank -> sorted index array.
// rank(s) = #{s' : key[s'] > key[s]} + #{s' < s : key[s'] == key[s]}
// sidx[ba][rank] = s   (rank is also the inverse permutation).
// ---------------------------------------------------------------------------
__global__ void nlsa_rank(const float* __restrict__ hash,
                          int* __restrict__ sidx) {
    __shared__ float keys[256];
    const int tiles = SS / 256;            // 36
    int ba   = blockIdx.x / tiles;
    int tile = blockIdx.x % tiles;
    int s = tile * 256 + threadIdx.x;
    float myk = hash[ba * SS + s];

    int cnt = 0;
    for (int t = 0; t < tiles; ++t) {
        __syncthreads();
        keys[threadIdx.x] = hash[ba * SS + t * 256 + threadIdx.x];
        __syncthreads();
#pragma unroll 8
        for (int q = 0; q < 256; ++q) {
            float o = keys[q];
            int so = t * 256 + q;
            cnt += (o > myk) || ((o == myk) && (so < s));
        }
    }
    sidx[ba * SS + cnt] = s;
}

// ---------------------------------------------------------------------------
// WMMA row-major fragment loader (wave32 bf16 layout, CDNA5 ISA §7.12.2).
// A (16xK tile): lane&15 = row, lanes>=16 shift K base by +8;
//   VGPR pair p holds K = (p&3)*2 {+16 if p>=4}. B mirrors with lane&15 = N,
//   so a B-fragment is a row-major fragment of the transposed matrix.
// Contiguous K pairs -> compiler emits packed ds_load_b128.
// ---------------------------------------------------------------------------
__device__ __forceinline__ v16bf frag_rowmajor(const __bf16* M, int ldm,
                                               int row0, int k0, int lane) {
    const __bf16* p = M + (row0 + (lane & 15)) * ldm + k0 + ((lane >> 4) << 3);
    v16bf f;
#pragma unroll
    for (int e = 0; e < 16; ++e) {
        int q = e >> 1;
        int kk = ((q & 3) << 1) + (e & 1) + ((q >> 2) << 4);
        f[e] = p[kk];
    }
    return f;
}

// ---------------------------------------------------------------------------
// Kernel 3: per-bucket attention + mean-scatter.
// One workgroup per (b, a, bucket): 9 waves, wave w owns query rows 16w..16w+15.
// ---------------------------------------------------------------------------
__global__ void __launch_bounds__(288)
nlsa_attn(const float* __restrict__ in,
          const int* __restrict__ sidx,
          float* __restrict__ out) {
    __shared__ __bf16 Xb[KB * CC];        // bucket features, row-major [i][c]
    __shared__ __bf16 Xbt[CC * KPAD];     // transposed copy [c][i], i>=144 zero
    __shared__ __bf16 Pm[KB * KPAD];      // softmax probs, cols 144..159 zero
    __shared__ int    sl[KB];             // gathered sequence positions

    int n = blockIdx.x % NB;
    int a = (blockIdx.x / NB) % AA;
    int b = blockIdx.x / (NB * AA);
    int ba = b * AA + a;

    int tid  = threadIdx.x;
    int lane = tid & 31;
    int w    = tid >> 5;                  // wave id 0..8 = query tile

    // --- phase 1: gather bucket rows -> bf16 LDS (both layouts); zero pads -
    for (int i = tid; i < KB; i += 288) sl[i] = sidx[ba * SS + n * KB + i];
    __syncthreads();
    for (int e = tid; e < KB * CC; e += 288) {
        int c = e / KB, i = e % KB;       // i varies fastest across threads
        __bf16 v = (__bf16)in[((b * CC) + c) * SS + sl[i]];
        Xb[i * CC + c] = v;
        Xbt[c * KPAD + i] = v;
    }
    for (int e = tid; e < CC * (KPAD - KB); e += 288) {
        int c = e / (KPAD - KB), i = KB + e % (KPAD - KB);
        Xbt[c * KPAD + i] = (__bf16)0.0f;
    }
    for (int e = tid; e < KB * KPAD; e += 288)
        Pm[e] = (__bf16)0.0f;
    __syncthreads();

    // --- phase 2: scores tile row: S[16 x 144] = Q_tile @ K^T --------------
    v8f acc[9];
#pragma unroll
    for (int j = 0; j < 9; ++j)
#pragma unroll
        for (int r = 0; r < 8; ++r) acc[j][r] = 0.0f;

    for (int kc = 0; kc < CC / 32; ++kc) {            // 8 K-chunks
        v16bf af = frag_rowmajor(Xb, CC, 16 * w, 32 * kc, lane);
#pragma unroll
        for (int j = 0; j < 9; ++j) {
            v16bf bf_ = frag_rowmajor(Xb, CC, 16 * j, 32 * kc, lane);
            acc[j] = __builtin_amdgcn_wmma_f32_16x16x32_bf16(
                false, af, false, bf_, (short)0, acc[j], false, false);
        }
    }

    // --- phase 3: softmax per row, write probs (bf16) to Pm ----------------
    const float scale = 0.0625f;                      // 1/sqrt(256)
#pragma unroll
    for (int r = 0; r < 8; ++r) {
        float mx = acc[0][r];
#pragma unroll
        for (int j = 1; j < 9; ++j) mx = fmaxf(mx, acc[j][r]);
#pragma unroll
        for (int m = 1; m <= 8; m <<= 1)              // reduce within 16-lane half
            mx = fmaxf(mx, __shfl_xor(mx, m, 32));
        float ex[9], sum = 0.0f;
#pragma unroll
        for (int j = 0; j < 9; ++j) {
            ex[j] = __expf((acc[j][r] - mx) * scale);
            sum += ex[j];
        }
#pragma unroll
        for (int m = 1; m <= 8; m <<= 1)
            sum += __shfl_xor(sum, m, 32);
        float inv = 1.0f / sum;
        int row = 16 * w + r + ((lane >> 4) << 3);
#pragma unroll
        for (int j = 0; j < 9; ++j)
            Pm[row * KPAD + 16 * j + (lane & 15)] = (__bf16)(ex[j] * inv);
    }
    // wave only reads its own Pm rows; DS-counter dependency handled by compiler

    // --- phase 4: O[16 x 256] = P[16 x 160] @ V[160 x 256]; atomic scatter -
    for (int half = 0; half < 2; ++half) {
        v8f o[8];
#pragma unroll
        for (int t = 0; t < 8; ++t)
#pragma unroll
            for (int r = 0; r < 8; ++r) o[t][r] = 0.0f;

        for (int kc = 0; kc < KPAD / 32; ++kc) {      // 5 K-chunks
            v16bf af = frag_rowmajor(Pm, KPAD, 16 * w, 32 * kc, lane);
#pragma unroll
            for (int t = 0; t < 8; ++t) {
                // B[K=key][N=c] == row-major fragment of Xbt[c][key]
                v16bf bf_ = frag_rowmajor(Xbt, KPAD,
                                          half * 128 + 16 * t, 32 * kc, lane);
                o[t] = __builtin_amdgcn_wmma_f32_16x16x32_bf16(
                    false, af, false, bf_, (short)0, o[t], false, false);
            }
        }
#pragma unroll
        for (int t = 0; t < 8; ++t) {
#pragma unroll
            for (int r = 0; r < 8; ++r) {
                int row = 16 * w + r + ((lane >> 4) << 3);
                int col = half * 128 + 16 * t + (lane & 15);
                int s = sl[row];
                atomicAdd(&out[((size_t)b * SS + s) * CC + col],
                          o[t][r] * (1.0f / AA));
            }
        }
    }
}

// ---------------------------------------------------------------------------
extern "C" void kernel_launch(void* const* d_in, const int* in_sizes, int n_in,
                              void* d_out, int out_size, void* d_ws, size_t ws_size,
                              hipStream_t stream) {
    (void)in_sizes; (void)n_in; (void)ws_size;
    const float* in = (const float*)d_in[0];          // [B, C, H, W] fp32
    const float* rm = (const float*)d_in[1];          // [A, C, 64]   fp32
    float* out = (float*)d_out;                       // [B, S, C]    fp32

    float* hash = (float*)d_ws;                                   // B*A*S f32
    int*   sidx = (int*)((char*)d_ws + (size_t)BB * AA * SS * 4); // B*A*S i32

    nlsa_zero<<<(out_size + 255) / 256, 256, 0, stream>>>(out, out_size);

    int hb = BB * AA * (SS / 256);                    // 1152 blocks
    nlsa_hash<<<hb, 256, 0, stream>>>(in, rm, hash);
    nlsa_rank<<<hb, 256, 0, stream>>>(hash, sidx);

    nlsa_attn<<<BB * AA * NB, 288, 0, stream>>>(in, sidx, out);
}
